// LSTM_69595650065036
// MI455X (gfx1250) — compile-verified, matrix operand order checked by reading, not run on hardware
//
#include <hip/hip_runtime.h>
#include <hip/hip_bf16.h>

// ---------------------------------------------------------------------------
// LSTM for MI455X (gfx1250, wave32, WMMA bf16 16x16x32, f32 accumulate)
//
// Shapes: seq=512, batch=32, input=H=1024, gates 4H=4096, layers=2 (identical).
// Phase 1: pack X and weights into WMMA fragment-ready bf16 order.
// Phase 2: big GEMM  x_gates = X * Wih^T + bias   (M=16384,N=4096,K=1024),
//          output stored directly in WMMA C/D accumulator order.
// Phase 3: persistent 32-block kernel, 1 global barrier per timestep.
//          Block b owns columns {g*1024 + b*32 .. +31} for all 4 gates, so the
//          f/i/g/o pointwise for hidden cols [b*32,b*32+32) is block-local.
//          New h is written bf16 in A-fragment order (K-tile kt=b) into a
//          double-buffered global hpack; barrier; next step every block stages
//          hpack (64 KB) into LDS via GLOBAL_LOAD_ASYNC_TO_LDS (ASYNCcnt) and
//          feeds WMMA A operands from ds_load_b128.
// ---------------------------------------------------------------------------

typedef __attribute__((ext_vector_type(16))) __bf16 v16bf;
typedef __attribute__((ext_vector_type(8)))  float  v8f;
typedef __attribute__((ext_vector_type(4)))  int    v4i;

#define SEQ   512
#define BATCH 32
#define HID   1024
#define NG    4096              // 4*HID
#define NT    (NG / 16)         // 256 n-tiles
#define KT    (HID / 32)        // 32  k-tiles
#define GMT   (SEQ * BATCH / 16)// 1024 global m-tiles (m = t*32 + batch_row)

// ---- workspace layout (bytes) ---------------------------------------------
#define XG_BYTES    ((size_t)SEQ * 2 * NT * 256 * 4)    // 256 MB  fp32, C/D order
#define PA_BYTES    ((size_t)GMT * KT * 32 * 16 * 2)    //  32 MB  bf16 A-frags of X
#define PB_BYTES    ((size_t)KT * NT * 32 * 16 * 2)     //   8 MB  bf16 B-frags
#define HPACK_BYTES ((size_t)2 * 2 * KT * 32 * 16 * 2)  // 128 KB  double-buffered h
#define HPACK_HALF  ((size_t)2 * KT * 32 * 16)          // elements per buffer

#if defined(__HIP_DEVICE_COMPILE__) &&                                   \
    __has_builtin(__builtin_amdgcn_global_load_async_to_lds_b128) &&     \
    __has_builtin(__builtin_amdgcn_s_wait_asynccnt)
#define USE_ASYNC_LDS 1
// Builtin signature (probe-confirmed by diagnostic): non-const int4* in as(1)
// for the global source, int4* in as(3) for the LDS destination, imm offset,
// imm cpol:  "vV4i*1V4i*3IiIi"
typedef __attribute__((address_space(1))) v4i glob_v4i;
typedef __attribute__((address_space(3))) v4i lds_v4i;
#else
#define USE_ASYNC_LDS 0
#endif

// Fragment index maps (CDNA5 ISA 7.12.2, 16-bit operands, wave32):
//  A (16x32): lane l -> row m = l&15 ; halves 0..7  -> K = kb..kb+7,
//             halves 8..15 -> K = kb+16..kb+23, kb = (l>=16)?8:0.
//  B (32x16): same indexing with N in place of M (row-of-B striped per lane).
//  C/D f32  : lane l -> col n = l&15 ; vgpr r -> row m = r + 8*(l>=16).

__device__ __forceinline__ float sigf(float x) {
  return __builtin_amdgcn_rcpf(1.0f + __expf(-x));
}
__device__ __forceinline__ float tanh_fast(float x) {
  float e = __expf(-2.0f * x);
  return (1.0f - e) * __builtin_amdgcn_rcpf(1.0f + e);
}

// ---- Phase 1a: pack input X (fp32 [16384][1024]) into bf16 A fragments -----
__global__ __launch_bounds__(256) void pack_a_kernel(const float* __restrict__ x,
                                                     __bf16* __restrict__ pa) {
  int id   = blockIdx.x * 256 + threadIdx.x;  // GMT*KT*32 = 1,048,576 threads
  int lane = id & 31;
  int kt   = (id >> 5) & 31;
  int gmt  = id >> 10;
  if (gmt >= GMT) return;
  int m  = gmt * 16 + (lane & 15);
  int kb = kt * 32 + ((lane >> 4) & 1) * 8;
  const float* src = x + (size_t)m * HID + kb;
  v16bf f;
#pragma unroll
  for (int i = 0; i < 8; ++i) f[i] = (__bf16)src[i];
#pragma unroll
  for (int i = 0; i < 8; ++i) f[8 + i] = (__bf16)src[16 + i];
  *(v16bf*)(pa + ((size_t)(gmt * KT + kt) * 32 + lane) * 16) = f;
}

// ---- Phase 1b: pack W (fp32 [4096][1024], row-major) into bf16 B fragments -
// B[k][n] = W[n][k]  (GEMM computes X * W^T)
__global__ __launch_bounds__(256) void pack_b_kernel(const float* __restrict__ w,
                                                     __bf16* __restrict__ pb) {
  int id   = blockIdx.x * 256 + threadIdx.x;  // KT*NT*32 = 262,144 threads
  int lane = id & 31;
  int nt   = (id >> 5) & 255;
  int kt   = id >> 13;
  if (kt >= KT) return;
  int n  = nt * 16 + (lane & 15);
  int kb = kt * 32 + ((lane >> 4) & 1) * 8;
  const float* src = w + (size_t)n * HID + kb;
  v16bf f;
#pragma unroll
  for (int i = 0; i < 8; ++i) f[i] = (__bf16)src[i];
#pragma unroll
  for (int i = 0; i < 8; ++i) f[8 + i] = (__bf16)src[16 + i];
  *(v16bf*)(pb + ((size_t)(kt * NT + nt) * 32 + lane) * 16) = f;
}

// ---- Phase 2: x_gates GEMM, output in C/D accumulator order ----------------
// grid 4096 blocks * 256 thr; block: 2 m-tiles x 32 n-tiles; wave: 2 mt x 4 nt
__global__ __launch_bounds__(256) void xgates_gemm_kernel(
    const __bf16* __restrict__ pa, const __bf16* __restrict__ pb,
    const float* __restrict__ bih, const float* __restrict__ bhh,
    float* __restrict__ xg) {
  int tid = threadIdx.x, lane = tid & 31, wave = tid >> 5;
  int bm = blockIdx.x >> 3;           // 0..511
  int bn = blockIdx.x & 7;            // 0..7
  int gmt0 = bm * 2;
  int nt0  = bn * 32 + wave * 4;

  v8f acc[4][2];
#pragma unroll
  for (int j = 0; j < 4; ++j) { acc[j][0] = {}; acc[j][1] = {}; }

  const __bf16* pa0 = pa + ((size_t)(gmt0 * KT) * 32 + lane) * 16;
  const __bf16* pa1 = pa + ((size_t)((gmt0 + 1) * KT) * 32 + lane) * 16;

  for (int kt = 0; kt < KT; ++kt) {
    v16bf a0 = *(const v16bf*)(pa0 + (size_t)kt * 32 * 16);
    v16bf a1 = *(const v16bf*)(pa1 + (size_t)kt * 32 * 16);
    if (kt + 1 < KT)
      __builtin_prefetch(pb + ((size_t)((kt + 1) * NT + nt0) * 32 + lane) * 16, 0, 2);
#pragma unroll
    for (int j = 0; j < 4; ++j) {
      v16bf b = *(const v16bf*)(pb + ((size_t)(kt * NT + nt0 + j) * 32 + lane) * 16);
      acc[j][0] = __builtin_amdgcn_wmma_f32_16x16x32_bf16(
          false, a0, false, b, (short)0, acc[j][0], false, false);
      acc[j][1] = __builtin_amdgcn_wmma_f32_16x16x32_bf16(
          false, a1, false, b, (short)0, acc[j][1], false, false);
    }
  }
#pragma unroll
  for (int j = 0; j < 4; ++j) {
    int n = (nt0 + j) * 16 + (lane & 15);
    float bv = bih[n] + bhh[n];
#pragma unroll
    for (int r = 0; r < 8; ++r) { acc[j][0][r] += bv; acc[j][1][r] += bv; }
#pragma unroll
    for (int mt = 0; mt < 2; ++mt)
      *(v8f*)(xg + ((size_t)(gmt0 + mt) * NT + (nt0 + j)) * 256 + lane * 8) =
          acc[j][mt];
  }
}

// ---- Phase 3: persistent recurrence, 32 blocks, 1 barrier per step ---------
__global__ __launch_bounds__(256) void lstm_recurrent_kernel(
    const float* __restrict__ xg, const __bf16* __restrict__ pwhh,
    __bf16* __restrict__ hpack, unsigned* __restrict__ bar,
    float* __restrict__ out) {
  __shared__ float gbuf[4][2][2][32][8];   // [gate][nsub][mtile][lane][r] 16 KB
  __shared__ float cbuf[32][32];           // cell-state slice, 4 KB
  __shared__ __bf16 habuf[2 * KT * 32 * 16];  // h A-fragments, 64 KB (<=320KB/WGP)

  int tid = threadIdx.x, lane = tid & 31, wave = tid >> 5;
  int blk  = blockIdx.x;                  // 0..31
  int gate = wave >> 1, sub = wave & 1;
  int ntile = gate * 64 + blk * 2 + sub;  // n0 = gate*1024 + blk*32 + sub*16

  for (int e = tid; e < 32 * 32; e += 256) ((float*)cbuf)[e] = 0.0f;
  __syncthreads();

  const size_t L = (size_t)SEQ * BATCH * HID;  // one layer of h_n / c_n

  for (int t = 0; t < SEQ; ++t) {
    const __bf16* hrd = hpack + (size_t)(t & 1) * HPACK_HALF;        // read buf
    __bf16*       hwr = hpack + (size_t)((t + 1) & 1) * HPACK_HALF;  // write buf

    // ---- stage h A-fragments into LDS (64 KB = 32768 bf16), 256 B/thread --
    {
      const __bf16* g = hrd + (size_t)tid * 8;   // 16 B granules
      __bf16*       l = habuf + (size_t)tid * 8;
#if USE_ASYNC_LDS
#pragma unroll
      for (int i = 0; i < 16; ++i) {             // 16 x b128 per thread
        __bf16* gs = const_cast<__bf16*>(g + (size_t)i * 2048);
        __bf16* ls = l + (size_t)i * 2048;
        __builtin_amdgcn_global_load_async_to_lds_b128(
            (glob_v4i*)gs, (lds_v4i*)ls, 0, 0);
      }
      __builtin_amdgcn_s_wait_asynccnt(0);
#else
#pragma unroll
      for (int i = 0; i < 8; ++i)                // 8 x 32 B per thread
        *(v16bf*)(l + (size_t)i * 4096) = *(const v16bf*)(g + (size_t)i * 4096);
#endif
    }
    __syncthreads();

    v8f acc0 = {}, acc1 = {};
#pragma unroll 4
    for (int kt = 0; kt < KT; ++kt) {
      v16bf a0 = *(const v16bf*)(habuf + ((size_t)(0 * KT + kt) * 32 + lane) * 16);
      v16bf a1 = *(const v16bf*)(habuf + ((size_t)(1 * KT + kt) * 32 + lane) * 16);
      if (kt + 1 < KT)
        __builtin_prefetch(
            pwhh + ((size_t)((kt + 1) * NT + ntile) * 32 + lane) * 16, 0, 2);
      v16bf b = *(const v16bf*)(pwhh + ((size_t)(kt * NT + ntile) * 32 + lane) * 16);
      acc0 = __builtin_amdgcn_wmma_f32_16x16x32_bf16(
          false, a0, false, b, (short)0, acc0, false, false);
      acc1 = __builtin_amdgcn_wmma_f32_16x16x32_bf16(
          false, a1, false, b, (short)0, acc1, false, false);
    }
    // add precomputed input gates (already includes bias), same C/D layout
    v8f x0 = *(const v8f*)(xg + ((size_t)(t * 2 + 0) * NT + ntile) * 256 + lane * 8);
    v8f x1 = *(const v8f*)(xg + ((size_t)(t * 2 + 1) * NT + ntile) * 256 + lane * 8);
#pragma unroll
    for (int r = 0; r < 8; ++r) {
      gbuf[gate][sub][0][lane][r] = acc0[r] + x0[r];
      gbuf[gate][sub][1][lane][r] = acc1[r] + x1[r];
    }
    __syncthreads();

    // pointwise: 32 rows x 32 local cols = 1024 elems, 4 per thread
#pragma unroll
    for (int e = 0; e < 4; ++e) {
      int idx = tid + 256 * e;
      int m = idx >> 5, jj = idx & 31;
      int mt = m >> 4, r = m & 7, hi = (m >> 3) & 1;
      int subp = jj >> 4, ln = (jj & 15) + 16 * hi;
      float fg = gbuf[0][subp][mt][ln][r];
      float ig = gbuf[1][subp][mt][ln][r];
      float gg = gbuf[2][subp][mt][ln][r];
      float og = gbuf[3][subp][mt][ln][r];
      float c  = cbuf[m][jj];
      float cn = sigf(fg) * c + sigf(ig) * tanh_fast(gg);
      float h  = sigf(og) * tanh_fast(cn);
      cbuf[m][jj] = cn;

      size_t j    = (size_t)blk * 32 + jj;
      size_t base = ((size_t)t * BATCH + m) * HID + j;
      out[base]         = h;   // h_n layer 0
      out[L + base]     = h;   // h_n layer 1 (layers are identical)
      out[2 * L + base] = cn;  // c_n layer 0
      out[3 * L + base] = cn;  // c_n layer 1

      // write new h in A-fragment order; block b fills K-tile kt = b
      int lane2 = (m & 15) + 16 * ((jj >> 3) & 1);
      int half  = (jj & 7) + 8 * ((jj >> 4) & 1);
      hwr[((size_t)(mt * KT + blk) * 32 + lane2) * 16 + half] = (__bf16)h;
    }
    __threadfence();
    __syncthreads();

    if (t < SEQ - 1) {
      if (tid == 0) {
        __hip_atomic_fetch_add(bar, 1u, __ATOMIC_RELEASE, __HIP_MEMORY_SCOPE_AGENT);
        unsigned target = 32u * (unsigned)(t + 1);
        while (__hip_atomic_load(bar, __ATOMIC_ACQUIRE, __HIP_MEMORY_SCOPE_AGENT) <
               target)
          __builtin_amdgcn_s_sleep(2);
      }
      __syncthreads();
    }
  }
}

// ---------------------------------------------------------------------------
extern "C" void kernel_launch(void* const* d_in, const int* in_sizes, int n_in,
                              void* d_out, int out_size, void* d_ws, size_t ws_size,
                              hipStream_t stream) {
  (void)in_sizes; (void)n_in; (void)out_size; (void)ws_size;
  const float* x   = (const float*)d_in[0];  // (512,32,1024)
  const float* wih = (const float*)d_in[1];  // (4096,1024)
  const float* whh = (const float*)d_in[2];  // (4096,1024)
  const float* bih = (const float*)d_in[3];  // (4096,)
  const float* bhh = (const float*)d_in[4];  // (4096,)
  float* out = (float*)d_out;                // h_n(2 layers) then c_n(2 layers)

  char* ws = (char*)d_ws;
  float*  xg    = (float*)ws;
  __bf16* pa    = (__bf16*)(ws + XG_BYTES);
  __bf16* pbih  = (__bf16*)(ws + XG_BYTES + PA_BYTES);
  __bf16* pbhh  = (__bf16*)(ws + XG_BYTES + PA_BYTES + PB_BYTES);
  __bf16* hpack = (__bf16*)(ws + XG_BYTES + PA_BYTES + 2 * PB_BYTES);
  unsigned* bar = (unsigned*)(ws + XG_BYTES + PA_BYTES + 2 * PB_BYTES + HPACK_BYTES);

  // zero h0 double-buffer + global barrier counter (graph-capturable memset)
  (void)hipMemsetAsync(hpack, 0, HPACK_BYTES + 256, stream);

  pack_a_kernel<<<4096, 256, 0, stream>>>(x, pa);
  pack_b_kernel<<<1024, 256, 0, stream>>>(wih, pbih);
  pack_b_kernel<<<1024, 256, 0, stream>>>(whh, pbhh);
  xgates_gemm_kernel<<<4096, 256, 0, stream>>>(pa, pbih, bih, bhh, xg);
  lstm_recurrent_kernel<<<32, 256, 0, stream>>>(xg, pbhh, hpack, bar, out);
}
